// SRU_19490561589968
// MI455X (gfx1250) — matrix operand order repaired and checked by reading, not compiled, not run on hardware
//
#include <hip/hip_runtime.h>
#include <hip/hip_bf16.h>

// ---------------------------------------------------------------------------
// SRU forward for MI455X (gfx1250, wave32).
//   Phase 1: ux = x @ W   via split-bf16 (hi+lo) emulated-fp32 WMMA GEMM.
//   Phase 2: sequential non-linear scan over time (cannot be parallelized in t).
// ---------------------------------------------------------------------------

typedef __attribute__((ext_vector_type(16))) __bf16 v16bf;
typedef __attribute__((ext_vector_type(8)))  __bf16 v8bf;
typedef __attribute__((ext_vector_type(4)))  __bf16 v4bf;
typedef __attribute__((ext_vector_type(8)))  float  v8f;

#define SRU_B 8
#define SRU_S 2048
#define SRU_H 1024
#define GM (SRU_B * SRU_S)   // 16384 rows
#define GK (SRU_H)           // 1024
#define GN (3 * SRU_H)       // 3072
#define TK 32                // K chunk per LDS stage (one bf16 WMMA depth)
#define LDP 40               // padded LDS row stride (bf16 elems): 80B rows

__device__ __forceinline__ v16bf cat8(v8bf lo, v8bf hi) {
    return __builtin_shufflevector(lo, hi, 0, 1, 2, 3, 4, 5, 6, 7,
                                           8, 9, 10, 11, 12, 13, 14, 15);
}

// 128x128 block tile, 8 waves as 4(row) x 2(col), each wave owns 32x64.
__global__ __launch_bounds__(256)
void sru_gemm_kernel(const float* __restrict__ X,   // [GM][GK]
                     const float* __restrict__ W,   // [GK][GN]
                     float* __restrict__ UX)        // [GM][GN]
{
    __shared__ __align__(16) __bf16 Ah[128][LDP];  // A hi, row-major [m][k]
    __shared__ __align__(16) __bf16 Al[128][LDP];  // A lo
    __shared__ __align__(16) __bf16 Bh[128][LDP];  // B hi, transposed [n][k]
    __shared__ __align__(16) __bf16 Bl[128][LDP];  // B lo

    const int tid  = threadIdx.x;
    const int lane = tid & 31;
    const int wid  = tid >> 5;       // 0..7
    const int wrow = wid & 3;        // 4 wave rows
    const int wcol = wid >> 2;       // 2 wave cols
    const int l16  = lane & 15;
    const int half = lane >> 4;      // lane half selects K sub-range

    const int m0 = blockIdx.y * 128;
    const int n0 = blockIdx.x * 128;

    // global loader mapping: A -> 128x32 floats, B -> 32x128 floats, 16/thread
    const int a_c4 = tid & 7;        // float4 column within 32-wide K chunk
    const int a_r  = tid >> 3;       // 0..31 (x4 passes -> 128 rows)
    const int b_n4 = tid & 31;       // float4 column within 128-wide N tile
    const int b_k  = tid >> 5;       // 0..7  (x4 passes -> 32 K rows)

    v8f acc[2][4];
#pragma unroll
    for (int i = 0; i < 2; ++i)
#pragma unroll
        for (int j = 0; j < 4; ++j)
            acc[i][j] = (v8f)(0.0f);

    for (int kc = 0; kc < GK; kc += TK) {
        // ---- prefetch next K chunk into L2 (global_prefetch_b8 path) ----
        if (kc + TK < GK) {
            __builtin_prefetch(&X[(size_t)(m0 + a_r) * GK + (kc + TK) + a_c4 * 4], 0, 0);
            __builtin_prefetch(&W[(size_t)(kc + TK + b_k) * GN + n0 + b_n4 * 4], 0, 0);
        }

        // ---- stage A tile (128x32 f32) as bf16 hi/lo ----
#pragma unroll
        for (int p = 0; p < 4; ++p) {
            const int r = a_r + p * 32;
            const float4 v = *(const float4*)&X[(size_t)(m0 + r) * GK + kc + a_c4 * 4];
            const float vv[4] = {v.x, v.y, v.z, v.w};
            v4bf h4, l4;
#pragma unroll
            for (int e = 0; e < 4; ++e) {
                const __bf16 h = (__bf16)vv[e];
                h4[e] = h;
                l4[e] = (__bf16)(vv[e] - (float)h);
            }
            *(v4bf*)&Ah[r][a_c4 * 4] = h4;
            *(v4bf*)&Al[r][a_c4 * 4] = l4;
        }

        // ---- stage B tile (32x128 f32) transposed [n][k] as bf16 hi/lo ----
#pragma unroll
        for (int p = 0; p < 4; ++p) {
            const int k = b_k + p * 8;
            const float4 v = *(const float4*)&W[(size_t)(kc + k) * GN + n0 + b_n4 * 4];
            const float vv[4] = {v.x, v.y, v.z, v.w};
#pragma unroll
            for (int e = 0; e < 4; ++e) {
                const __bf16 h = (__bf16)vv[e];
                Bh[b_n4 * 4 + e][k] = h;
                Bl[b_n4 * 4 + e][k] = (__bf16)(vv[e] - (float)h);
            }
        }
        __syncthreads();

        // ---- fragment loads per ISA operand layouts ----
        // A 16x32 (16-bit): lane<16 holds K{0..7,16..23}; lane>=16 K{8..15,24..31}
        v16bf afh[2], afl[2];
#pragma unroll
        for (int i = 0; i < 2; ++i) {
            const int row = wrow * 32 + i * 16 + l16;
            const int k0 = half * 8;
            const int k1 = 16 + half * 8;
            afh[i] = cat8(*(const v8bf*)&Ah[row][k0], *(const v8bf*)&Ah[row][k1]);
            afl[i] = cat8(*(const v8bf*)&Al[row][k0], *(const v8bf*)&Al[row][k1]);
        }
        // B 32x16 (16-bit): lane<16 holds K{0..7,16..23} of col N=l16; lane>=16 K{8..15,24..31}
        v16bf bfh[4], bfl[4];
#pragma unroll
        for (int j = 0; j < 4; ++j) {
            const int n  = wcol * 64 + j * 16 + l16;
            const int kb = half * 8;
            bfh[j] = cat8(*(const v8bf*)&Bh[n][kb], *(const v8bf*)&Bh[n][kb + 16]);
            bfl[j] = cat8(*(const v8bf*)&Bl[n][kb], *(const v8bf*)&Bl[n][kb + 16]);
        }

        // ---- split-bf16 fp32 emulation: hi*hi + hi*lo + lo*hi ----
#pragma unroll
        for (int i = 0; i < 2; ++i) {
#pragma unroll
            for (int j = 0; j < 4; ++j) {
                acc[i][j] = __builtin_amdgcn_wmma_f32_16x16x32_bf16(
                    false, afh[i], false, bfh[j], (short)0, acc[i][j], false, false);
                acc[i][j] = __builtin_amdgcn_wmma_f32_16x16x32_bf16(
                    false, afh[i], false, bfl[j], (short)0, acc[i][j], false, false);
                acc[i][j] = __builtin_amdgcn_wmma_f32_16x16x32_bf16(
                    false, afl[i], false, bfh[j], (short)0, acc[i][j], false, false);
            }
        }
        __syncthreads();
    }

    // ---- store C/D (VGPR r: lanes 0-15 -> M=r, lanes 16-31 -> M=8+r) ----
#pragma unroll
    for (int i = 0; i < 2; ++i) {
#pragma unroll
        for (int j = 0; j < 4; ++j) {
            const int gcol = n0 + wcol * 64 + j * 16 + l16;
#pragma unroll
            for (int r = 0; r < 8; ++r) {
                const int grow = m0 + wrow * 32 + i * 16 + half * 8 + r;
                UX[(size_t)grow * GN + gcol] = acc[i][j][r];
            }
        }
    }
}

// One thread per (b,h) chain; consecutive threads -> consecutive h (coalesced).
__global__ __launch_bounds__(256)
void sru_scan_kernel(const float* __restrict__ X,    // [B][S][H]
                     const float* __restrict__ UX,   // [B*S][3H]
                     const float* __restrict__ vf,
                     const float* __restrict__ vr,
                     const float* __restrict__ bf,
                     const float* __restrict__ br,
                     float* __restrict__ Y,          // [B][S][H]
                     float* __restrict__ Cf)         // [B][H]
{
    const int g = blockIdx.x * blockDim.x + threadIdx.x;  // 0 .. B*H-1
    const int b = g >> 10;        // / H
    const int h = g & (SRU_H - 1);

    const float vfh = vf[h];
    const float vrh = vr[h];
    const float bfh = bf[h];
    const float brh = br[h];

    const float* xrow = X + (size_t)b * SRU_S * SRU_H + h;
    const float* urow = UX + (size_t)b * SRU_S * GN + h;
    float*       yrow = Y + (size_t)b * SRU_S * SRU_H + h;

    float c = 0.0f;
    for (int t = 0; t < SRU_S; ++t) {
        if (t + 8 < SRU_S) {
            __builtin_prefetch(&urow[(size_t)(t + 8) * GN], 0, 0);
            __builtin_prefetch(&xrow[(size_t)(t + 8) * SRU_H], 0, 0);
        }
        const float xt = xrow[(size_t)t * SRU_H];
        const float xf = urow[(size_t)t * GN];
        const float xc = urow[(size_t)t * GN + SRU_H];
        const float xr = urow[(size_t)t * GN + 2 * SRU_H];

        const float ft = 1.0f / (1.0f + __expf(-(xf + bfh + vfh * c)));
        const float rt = 1.0f / (1.0f + __expf(-(xr + brh + vrh * c)));
        c = ft * c + (1.0f - ft) * xc;
        yrow[(size_t)t * SRU_H] = rt * c + (1.0f - rt) * xt;
    }
    Cf[g] = c;
}

extern "C" void kernel_launch(void* const* d_in, const int* in_sizes, int n_in,
                              void* d_out, int out_size, void* d_ws, size_t ws_size,
                              hipStream_t stream) {
    const float* X  = (const float*)d_in[0];  // [8,2048,1024]
    const float* W  = (const float*)d_in[1];  // [1024,3072]
    const float* vf = (const float*)d_in[2];  // [1024]
    const float* vr = (const float*)d_in[3];
    const float* bf = (const float*)d_in[4];
    const float* br = (const float*)d_in[5];

    float* UX = (float*)d_ws;                 // [16384][3072] = 192 MiB scratch
    float* Y  = (float*)d_out;                // [8,2048,1024]
    float* Cf = Y + (size_t)SRU_B * SRU_S * SRU_H;  // [8,1024]

    dim3 ggrid(GN / 128, GM / 128);           // (24, 128)
    sru_gemm_kernel<<<ggrid, 256, 0, stream>>>(X, W, UX);

    const int chains = SRU_B * SRU_H;         // 8192
    sru_scan_kernel<<<chains / 256, 256, 0, stream>>>(X, UX, vf, vr, bf, br, Y, Cf);
}
